// ROIAlignSpatialAttention_modi1_87780541596078
// MI455X (gfx1250) — matrix operand order repaired and checked by reading, not compiled, not run on hardware
//
#include <hip/hip_runtime.h>
#include <hip/hip_bf16.h>
#include <math.h>

// Problem constants (from reference)
#define B_    2
#define NROI  1024
#define WS_   5
#define CH    18
#define E_    25          // WS*WS
#define HID   450         // E*CH
#define NTOT  (B_*NROI)   // 2048

typedef __attribute__((ext_vector_type(16))) __bf16 v16bf;
typedef __attribute__((ext_vector_type(8)))  float  v8f;

#define AS1 __attribute__((address_space(1)))
#define AS3 __attribute__((address_space(3)))

#ifndef __has_builtin
#define __has_builtin(x) 0
#endif

#if __has_builtin(__builtin_amdgcn_global_load_async_to_lds_b32)
#define HAVE_ASYNC_LDS 1
#else
#define HAVE_ASYNC_LDS 0
#endif

static __device__ __forceinline__ int imin(int a, int b) { return a < b ? a : b; }

static __device__ __forceinline__ v8f vzero() {
  v8f z = {0.f,0.f,0.f,0.f,0.f,0.f,0.f,0.f};
  return z;
}

static __device__ __forceinline__ __bf16 to_bf(float f) { return (__bf16)f; }

static __device__ __forceinline__ v8f wmma_bf16(v16bf a, v16bf b, v8f c) {
  // D = A(16x32 bf16) x B(32x16 bf16) + C(16x16 f32)
  return __builtin_amdgcn_wmma_f32_16x16x32_bf16(false, a, false, b, (short)0, c, false, false);
}

// Async global->LDS copy (gfx1250 GLOBAL_LOAD_ASYNC_TO_LDS_B32, ASYNCcnt),
// with plain-copy fallback if the builtin is unavailable.
static __device__ __forceinline__ void async_copy_b32(const float* g, float* l) {
#if HAVE_ASYNC_LDS
  __builtin_amdgcn_global_load_async_to_lds_b32((AS1 int*)(void*)g,
                                                (AS3 int*)(void*)l, 0, 0);
#else
  *l = *g;
#endif
}

static __device__ __forceinline__ void wait_async0() {
#if __has_builtin(__builtin_amdgcn_s_wait_asynccnt)
  __builtin_amdgcn_s_wait_asynccnt(0);
#elif HAVE_ASYNC_LDS
  asm volatile("s_wait_asynccnt 0" ::: "memory");
#endif
}

// ---------------------------------------------------------------------------
// Fragment loaders
// ---------------------------------------------------------------------------
// Fast A loader: element (m,k) at p[m*mstride + k]; K contiguous; mstride*4 %16==0.
// Per-lane K set is two contiguous runs of 8 -> 4x ds_load_b128 + pack-cvt.
static __device__ __forceinline__ v16bf loadA_fast(const float* p, int mstride) {
  int lane = threadIdx.x & 31;
  int m = lane & 15, h = lane >> 4;
  const float* b0 = p + m*mstride + h*8;
  float4 r0 = *(const float4*)(b0);
  float4 r1 = *(const float4*)(b0 + 4);
  float4 s0 = *(const float4*)(b0 + 16);
  float4 s1 = *(const float4*)(b0 + 20);
  v16bf a;
  a[0]=to_bf(r0.x); a[1]=to_bf(r0.y); a[2]=to_bf(r0.z); a[3]=to_bf(r0.w);
  a[4]=to_bf(r1.x); a[5]=to_bf(r1.y); a[6]=to_bf(r1.z); a[7]=to_bf(r1.w);
  a[8]=to_bf(s0.x); a[9]=to_bf(s0.y); a[10]=to_bf(s0.z); a[11]=to_bf(s0.w);
  a[12]=to_bf(s1.x); a[13]=to_bf(s1.y); a[14]=to_bf(s1.z); a[15]=to_bf(s1.w);
  return a;
}

// Slow guarded A loader (used only for fc1 / unaligned rows).
static __device__ __forceinline__ v16bf loadA(const float* p, int ks, int ms, int kvalid) {
  int lane = threadIdx.x & 31;
  int m = lane & 15, h = lane >> 4;
  v16bf a;
#pragma unroll
  for (int v = 0; v < 8; ++v) {
    int k = ((v >> 2) << 4) + (h << 3) + ((v & 3) << 1);
    a[2*v]   = (k     < kvalid) ? to_bf(p[(size_t)k      * ks + (size_t)m * ms]) : to_bf(0.f);
    a[2*v+1] = ((k+1) < kvalid) ? to_bf(p[(size_t)(k+1)  * ks + (size_t)m * ms]) : to_bf(0.f);
  }
  return a;
}

// Packed B fragment: 32 lanes x 16 bf16 (32B per lane) per fragment -> 2x b128 loads.
static __device__ __forceinline__ v16bf loadBpack(const __bf16* pack, int frag) {
  return ((const v16bf*)pack)[(size_t)frag * 32 + (threadIdx.x & 31)];
}

// ---------------------------------------------------------------------------
// Weight repack kernels: write bf16 fragments in the exact WMMA-B VGPR layout.
// Layout: lane L holds n=L%16; VGPR v half p -> k=(v/4)*16+(L/16)*8+(v%4)*2+p.
// One 32-lane wave per fragment; each lane writes its 16 contiguous bf16.
// ---------------------------------------------------------------------------
__global__ __launch_bounds__(32)
void repack_conv1(const float* __restrict__ w1, int Cin, __bf16* __restrict__ dst)
{
  int f = blockIdx.x;            // f = (chunk*9 + tap)*2 + nt
  int nt  = f & 1;
  int tap = (f >> 1) % 9;
  int chunk = f / 18;
  int lane = threadIdx.x;
  int n = lane & 15, h = lane >> 4;
  __bf16* o = dst + (size_t)f * 512 + lane * 16;
#pragma unroll
  for (int v = 0; v < 8; ++v) {
#pragma unroll
    for (int p = 0; p < 2; ++p) {
      int k  = (v>>2)*16 + h*8 + (v&3)*2 + p;  // input channel within chunk
      int oc = nt*16 + n;               // output channel
      float val = (oc < CH) ? w1[((size_t)oc * Cin + chunk*32 + k) * 9 + tap] : 0.f;
      o[v*2 + p] = to_bf(val);
    }
  }
}

__global__ __launch_bounds__(32)
void repack_dense(const float* __restrict__ w, int N, int K, int ntiles,
                  __bf16* __restrict__ dst)
{
  int f = blockIdx.x;            // f = kc*ntiles + nt
  int nt = f % ntiles;
  int kc = f / ntiles;
  int lane = threadIdx.x;
  int n = lane & 15, h = lane >> 4;
  __bf16* o = dst + (size_t)f * 512 + lane * 16;
#pragma unroll
  for (int v = 0; v < 8; ++v) {
#pragma unroll
    for (int p = 0; p < 2; ++p) {
      int k  = kc*32 + (v>>2)*16 + h*8 + (v&3)*2 + p;
      int oc = nt*16 + n;
      float val = (oc < N && k < K) ? w[(size_t)oc * K + k] : 0.f;
      o[v*2 + p] = to_bf(val);
    }
  }
}

// ---------------------------------------------------------------------------
// Kernel 1: fused conv3x3(Cin->18) + BN + ReLU + conv1x1(18->18)
// One wave per (batch, row, 16-pixel strip). Implicit GEMM over 9 taps.
// Input tile staged to LDS with async-to-LDS; channel-contiguous layout so
// A fragments are 4x ds_load_b128.
// ---------------------------------------------------------------------------
__global__ __launch_bounds__(32)
void preconv_kernel(const float* __restrict__ in, int Cin, int H, int W,
                    const __bf16* __restrict__ w1p, const float* __restrict__ b1,
                    const float* __restrict__ gamma, const float* __restrict__ beta,
                    const float* __restrict__ mean, const float* __restrict__ var,
                    const __bf16* __restrict__ w2p, const float* __restrict__ b2,
                    float* __restrict__ out)
{
  __shared__ float tile[3*18*36];   // [row][col][channel], stride 36 (16B aligned)
  __shared__ float relu1[16*36];    // [pixel][channel], stride 36

  int lane = threadIdx.x & 31;
  int xblocks = (W + 15) >> 4;
  int bid = blockIdx.x;
  int xb = bid % xblocks;
  int y  = (bid / xblocks) % H;
  int b  = bid / (xblocks * H);
  int x0 = xb << 4;

  // Zero-fill once: out-of-bounds halo slots never get overwritten.
  for (int e = lane; e < 3*18*36; e += 32) tile[e] = 0.f;
  __syncthreads();

  v8f acc0 = vzero();
  v8f acc1 = vzero();

  for (int c0 = 0; c0 < Cin; c0 += 32) {
    __syncthreads();
    // Async-stage 32 channels x 3 rows x 18 cols into LDS.
    for (int pos = lane; pos < 54; pos += 32) {
      int ry = pos / 18, cx = pos - ry*18;
      int gy = y - 1 + ry, gx = x0 - 1 + cx;
      if (gy >= 0 && gy < H && gx >= 0 && gx < W) {
        const float* g = in + (((size_t)b * Cin + c0) * H + gy) * W + gx;
        float* l = tile + (ry*18 + cx)*36;
        size_t cs = (size_t)H * W;
#pragma unroll
        for (int cc = 0; cc < 32; ++cc)
          async_copy_b32(g + (size_t)cc * cs, l + cc);
      }
    }
    wait_async0();
    __syncthreads();

    int chunk = c0 >> 5;
#pragma unroll
    for (int tap = 0; tap < 9; ++tap) {
      int dy = tap / 3, dx = tap % 3;
      v16bf A = loadA_fast(tile + (dy*18 + dx)*36, 36);
      int fbase = (chunk*9 + tap)*2;
      acc0 = wmma_bf16(A, loadBpack(w1p, fbase),     acc0);
      acc1 = wmma_bf16(A, loadBpack(w1p, fbase + 1), acc1);
    }
  }

  // Bias + BN + ReLU, scatter C-layout into LDS [pixel][channel]
  int n  = lane & 15;
  int hh = lane >> 4;
#pragma unroll
  for (int t = 0; t < 2; ++t) {
    int ch = t*16 + n;
    float sc = 0.f, sh = 0.f, bb = 0.f, mu = 0.f;
    if (ch < CH) {
      sc = gamma[ch] * rsqrtf(var[ch] + 1e-3f);
      sh = beta[ch];
      bb = b1[ch];
      mu = mean[ch];
    }
    v8f a = (t == 0) ? acc0 : acc1;
#pragma unroll
    for (int r = 0; r < 8; ++r) {
      int m = hh*8 + r;
      float vconv = a[r] + bb;
      float vout  = (ch < CH) ? fmaxf((vconv - mu) * sc + sh, 0.f) : 0.f;
      relu1[m*36 + ch] = vout;   // channels 18..31 written as zero
    }
  }
  __syncthreads();

  // 1x1 conv 18->18 via one WMMA pair (K padded to 32 with zeros)
  v8f o0 = vzero(), o1 = vzero();
  v16bf A2 = loadA_fast(relu1, 36);
  o0 = wmma_bf16(A2, loadBpack(w2p, 0), o0);
  o1 = wmma_bf16(A2, loadBpack(w2p, 1), o1);

#pragma unroll
  for (int t = 0; t < 2; ++t) {
    int ch = t*16 + n;
    if (ch < CH) {
      float bb2 = b2[ch];
      v8f a = (t == 0) ? o0 : o1;
#pragma unroll
      for (int r = 0; r < 8; ++r) {
        int m = hh*8 + r;
        int x = x0 + m;
        if (x < W)
          out[(((size_t)b * CH + ch) * H + y) * W + x] = a[r] + bb2;
      }
    }
  }
}

// ---------------------------------------------------------------------------
// Kernel 2: ROI box generation + nearest-neighbor grid sample (scalar gather)
// ---------------------------------------------------------------------------
__global__ void sample_kernel(const float* __restrict__ rois,
                              const float* __restrict__ ffv,
                              const float* __restrict__ fbev,
                              float* __restrict__ sfv,
                              float* __restrict__ sbev)
{
  int idx = blockIdx.x * blockDim.x + threadIdx.x;
  if (idx >= NTOT * E_) return;
  int pt   = idx % E_;
  int nroi = idx / E_;
  int b = nroi / NROI;
  int r = nroi % NROI;
  const float* R = rois + ((size_t)b * NROI + r) * 7;
  float cx = R[0], cy = R[1], cz = R[2];
  float dx = R[3], dy = R[4], dz = R[5], th = R[6];
  float c = cosf(th), s = sinf(th);
  float wx = (dy*c + dx*s) * 0.5f;
  float wy = (dy*s + dx*c) * 0.5f;
  float bx1 = cx - wx, bx2 = cx + wx;
  float by1 = cy - wy, by2 = cy + wy;
  float bz1 = cz - dz*0.5f, bz2 = cz + dz*0.5f;

  int i = pt / WS_, j = pt % WS_;
  float ti = (float)i * 0.25f;
  float tj = (float)j * 0.25f;

  // FV: bbox = (z1, x1, z2, x2), feat (18, 64, 512)
  {
    float gz = bz1 + ti * (bz2 - bz1);  gz = (gz + 1.f) * 0.5f;
    float gx = bx1 + tj * (bx2 - bx1);  gx = (gx - 35.2f) / 35.2f;
    const int Hh = 64, Ww = 512;
    float ix = rintf((gz + 1.f) * 0.5f * (float)(Ww - 1));
    float iy = rintf((gx + 1.f) * 0.5f * (float)(Hh - 1));
    bool valid = (ix >= 0.f) && (ix <= (float)(Ww-1)) && (iy >= 0.f) && (iy <= (float)(Hh-1));
    int ixc = (int)fminf(fmaxf(ix, 0.f), (float)(Ww-1));
    int iyc = (int)fminf(fmaxf(iy, 0.f), (float)(Hh-1));
    for (int cc = 0; cc < CH; ++cc) {
      float v = valid ? ffv[(((size_t)b * CH + cc) * Hh + iyc) * Ww + ixc] : 0.f;
      sfv[(size_t)nroi * HID + cc * E_ + pt] = v;
    }
  }
  // BEV: bbox = (y1, x1, y2, x2), feat (18, 200, 176)
  {
    float gz = by1 + ti * (by2 - by1);  gz = gz / 40.f;
    float gx = bx1 + tj * (bx2 - bx1);  gx = (gx - 35.2f) / 35.2f;
    const int Hh = 200, Ww = 176;
    float ix = rintf((gz + 1.f) * 0.5f * (float)(Ww - 1));
    float iy = rintf((gx + 1.f) * 0.5f * (float)(Hh - 1));
    bool valid = (ix >= 0.f) && (ix <= (float)(Ww-1)) && (iy >= 0.f) && (iy <= (float)(Hh-1));
    int ixc = (int)fminf(fmaxf(ix, 0.f), (float)(Ww-1));
    int iyc = (int)fminf(fmaxf(iy, 0.f), (float)(Hh-1));
    for (int cc = 0; cc < CH; ++cc) {
      float v = valid ? fbev[(((size_t)b * CH + cc) * Hh + iyc) * Ww + ixc] : 0.f;
      sbev[(size_t)nroi * HID + cc * E_ + pt] = v;
    }
  }
}

// ---------------------------------------------------------------------------
// Kernel 3: per-ROI dual MHSA + 3x3 conv (36->18). One wave per ROI.
// ---------------------------------------------------------------------------
static __device__ void proj_store(const float* src, const __bf16* Wp, const float* bias,
                                  float* dst)
{
  int lane = threadIdx.x & 31;
  int n  = lane & 15;
  int hh = lane >> 4;
#pragma unroll
  for (int mt = 0; mt < 2; ++mt) {
    v16bf A = loadA_fast(src + mt*16*32, 32);
#pragma unroll
    for (int nt = 0; nt < 2; ++nt) {
      v8f acc = vzero();
      acc = wmma_bf16(A, loadBpack(Wp, nt), acc);
#pragma unroll
      for (int r = 0; r < 8; ++r) {
        int m   = mt*16 + hh*8 + r;
        int col = nt*16 + n;
        dst[m*32 + col] = (col < E_) ? (acc[r] + bias[col]) : 0.f;
      }
    }
  }
}

static __device__ void outproj_residual(const float* av, const __bf16* Wfp, const float* bf_,
                                        const float* f1, float* dst /* [CH*E_] */)
{
  int lane = threadIdx.x & 31;
  int n  = lane & 15;
  int hh = lane >> 4;
#pragma unroll
  for (int mt = 0; mt < 2; ++mt) {
    v16bf A = loadA_fast(av + mt*16*32, 32);
#pragma unroll
    for (int nt = 0; nt < 2; ++nt) {
      v8f acc = vzero();
      acc = wmma_bf16(A, loadBpack(Wfp, nt), acc);
#pragma unroll
      for (int r = 0; r < 8; ++r) {
        int m   = mt*16 + hh*8 + r;
        int col = nt*16 + n;
        if (m < CH && col < E_)
          dst[m*E_ + col] = acc[r] + bf_[col] + f1[m*32 + col];
      }
    }
  }
}

static __device__ void mhsa_dev(const float* f1, const float* f2,
                                const __bf16* Wqp, const float* bq,
                                const __bf16* Wkp, const float* bk,
                                const __bf16* Wvp, const float* bv,
                                const __bf16* Wfp, const float* bf_,
                                float* q, float* k, float* v, float* av,
                                float* dst)
{
  int lane = threadIdx.x & 31;
  proj_store(f2, Wqp, bq, q);   // q from feat2
  proj_store(f1, Wkp, bk, k);   // k from feat1
  proj_store(f2, Wvp, bv, v);   // v from feat2
  for (int e = lane; e < 32*32; e += 32) av[e] = 0.f;
  __syncthreads();

  const float scale = 0.44721359549995793f; // 1/sqrt(hd=5)
  for (int task = lane; task < WS_ * CH; task += 32) {
    int h = task / CH;
    int t = task % CH;
    float qv[5];
#pragma unroll
    for (int d = 0; d < 5; ++d) qv[d] = q[t*32 + h*5 + d];
    float sc[CH];
    float mx = -1e30f;
#pragma unroll
    for (int jj = 0; jj < CH; ++jj) {
      float sdot = 0.f;
#pragma unroll
      for (int d = 0; d < 5; ++d) sdot += qv[d] * k[jj*32 + h*5 + d];
      sdot *= scale;
      sc[jj] = sdot;
      mx = fmaxf(mx, sdot);
    }
    float sum = 0.f;
#pragma unroll
    for (int jj = 0; jj < CH; ++jj) { sc[jj] = expf(sc[jj] - mx); sum += sc[jj]; }
    float inv = 1.f / sum;
#pragma unroll
    for (int d = 0; d < 5; ++d) {
      float o = 0.f;
#pragma unroll
      for (int jj = 0; jj < CH; ++jj) o += sc[jj] * v[jj*32 + h*5 + d];
      av[t*32 + h*5 + d] = o * inv;
    }
  }
  __syncthreads();
  outproj_residual(av, Wfp, bf_, f1, dst);
  __syncthreads();
}

__global__ __launch_bounds__(32)
void classifier_kernel(const float* __restrict__ sbev, const float* __restrict__ sfv,
                       const __bf16* __restrict__ Wqp, const float* __restrict__ bq,
                       const __bf16* __restrict__ Wkp, const float* __restrict__ bk,
                       const __bf16* __restrict__ Wvp, const float* __restrict__ bv,
                       const __bf16* __restrict__ Wfp, const float* __restrict__ bf_,
                       const float* __restrict__ conv_w, const float* __restrict__ conv_b,
                       float* __restrict__ xflat)
{
  __shared__ float fb[32*32], ff[32*32];
  __shared__ float q[32*32], k[32*32], v[32*32], av[32*32];
  __shared__ float x1[CH*E_], x2[CH*E_];

  int lane = threadIdx.x;
  int roi  = blockIdx.x;

  for (int e = lane; e < 32*32; e += 32) {
    int row = e >> 5, col = e & 31;
    float vb = 0.f, vf = 0.f;
    if (row < CH && col < E_) {
      vb = sbev[(size_t)roi * HID + row * E_ + col];
      vf = sfv [(size_t)roi * HID + row * E_ + col];
    }
    fb[e] = vb;
    ff[e] = vf;
  }
  __syncthreads();

  mhsa_dev(fb, ff, Wqp, bq, Wkp, bk, Wvp, bv, Wfp, bf_, q, k, v, av, x1);
  mhsa_dev(ff, fb, Wqp, bq, Wkp, bk, Wvp, bv, Wfp, bf_, q, k, v, av, x2);

  // concat(x1,x2) -> 3x3 conv (36->18) over 5x5 -> xflat[roi][450]
  for (int oidx = lane; oidx < CH * E_; oidx += 32) {
    int o = oidx / E_, pix = oidx % E_;
    int i = pix / WS_, j = pix % WS_;
    float acc = conv_b[o];
    for (int ci = 0; ci < 2*CH; ++ci) {
      const float* src  = (ci < CH) ? (x1 + ci*E_) : (x2 + (ci - CH)*E_);
      const float* wrow = conv_w + ((size_t)o * 2*CH + ci) * 9;
#pragma unroll
      for (int dy2 = 0; dy2 < 3; ++dy2) {
        int ii = i + dy2 - 1;
        if (ii < 0 || ii >= WS_) continue;
#pragma unroll
        for (int dx2 = 0; dx2 < 3; ++dx2) {
          int jj = j + dx2 - 1;
          if (jj < 0 || jj >= WS_) continue;
          acc += src[ii*WS_ + jj] * wrow[dy2*3 + dx2];
        }
      }
    }
    xflat[(size_t)roi * HID + oidx] = acc;
  }
}

// ---------------------------------------------------------------------------
// Kernel 4: fc1 GEMM (2048 x 225 x 450) + ReLU via WMMA.
// ---------------------------------------------------------------------------
__global__ __launch_bounds__(32)
void fc1_kernel(const float* __restrict__ xflat,
                const __bf16* __restrict__ wp, const float* __restrict__ b,
                float* __restrict__ h)
{
  const int NT = 15;                 // ceil(225/16)
  int nt = blockIdx.x % NT;
  int mt = blockIdx.x / NT;
  int lane = threadIdx.x;
  int n  = lane & 15;
  int hh = lane >> 4;
  int n0 = nt * 16;
  int m0 = mt * 16;

  v8f acc = vzero();
  for (int kc = 0; kc < 15; ++kc) {
    int k0 = kc * 32;
    int kv = imin(32, HID - k0);
    v16bf A  = loadA(xflat + (size_t)m0 * HID + k0, 1, HID, kv);
    acc = wmma_bf16(A, loadBpack(wp, kc*NT + nt), acc);
  }

  int col = n0 + n;
  if (col < 225) {
    float bb = b[col];
#pragma unroll
    for (int r = 0; r < 8; ++r) {
      int m = m0 + hh*8 + r;
      h[(size_t)m * 225 + col] = fmaxf(acc[r] + bb, 0.f);
    }
  }
}

// ---------------------------------------------------------------------------
// Kernel 5: fc2 (225->3) + softmax + argmax. One thread per ROI.
// ---------------------------------------------------------------------------
__global__ void fc2_kernel(const float* __restrict__ h,
                           const float* __restrict__ w, const float* __restrict__ b,
                           float* __restrict__ out)
{
  int roi = blockIdx.x * blockDim.x + threadIdx.x;
  if (roi >= NTOT) return;
  const float* hr = h + (size_t)roi * 225;
  float s0 = b[0], s1 = b[1], s2 = b[2];
  for (int kk = 0; kk < 225; ++kk) {
    float x = hr[kk];
    s0 += x * w[kk];
    s1 += x * w[225 + kk];
    s2 += x * w[450 + kk];
  }
  float mx = fmaxf(s0, fmaxf(s1, s2));
  float e0 = expf(s0 - mx), e1 = expf(s1 - mx), e2 = expf(s2 - mx);
  float inv = 1.f / (e0 + e1 + e2);
  int pred = 0; float best = s0;
  if (s1 > best) { best = s1; pred = 1; }
  if (s2 > best) { best = s2; pred = 2; }
  out[roi] = (float)(pred + 1);
  out[NTOT + (size_t)roi*3 + 0] = e0 * inv;
  out[NTOT + (size_t)roi*3 + 1] = e1 * inv;
  out[NTOT + (size_t)roi*3 + 2] = e2 * inv;
}

// ---------------------------------------------------------------------------
extern "C" void kernel_launch(void* const* d_in, const int* in_sizes, int n_in,
                              void* d_out, int out_size, void* d_ws, size_t ws_size,
                              hipStream_t stream)
{
  const float* feat_fv  = (const float*)d_in[0];
  const float* feat_bev = (const float*)d_in[1];
  const float* rois     = (const float*)d_in[2];
  const float* fv_w1    = (const float*)d_in[3];
  const float* fv_b1    = (const float*)d_in[4];
  const float* fv_gamma = (const float*)d_in[5];
  const float* fv_beta  = (const float*)d_in[6];
  const float* fv_mean  = (const float*)d_in[7];
  const float* fv_var   = (const float*)d_in[8];
  const float* fv_w2    = (const float*)d_in[9];
  const float* fv_b2    = (const float*)d_in[10];
  const float* bev_w1   = (const float*)d_in[11];
  const float* bev_b1   = (const float*)d_in[12];
  const float* bev_gamma= (const float*)d_in[13];
  const float* bev_beta = (const float*)d_in[14];
  const float* bev_mean = (const float*)d_in[15];
  const float* bev_var  = (const float*)d_in[16];
  const float* bev_w2   = (const float*)d_in[17];
  const float* bev_b2   = (const float*)d_in[18];
  const float* Wq = (const float*)d_in[19];
  const float* bq = (const float*)d_in[20];
  const float* Wk = (const float*)d_in[21];
  const float* bk = (const float*)d_in[22];
  const float* Wv = (const float*)d_in[23];
  const float* bv = (const float*)d_in[24];
  const float* Wf = (const float*)d_in[25];
  const float* bf_ = (const float*)d_in[26];
  const float* conv_w = (const float*)d_in[27];
  const float* conv_b = (const float*)d_in[28];
  const float* fc1_w  = (const float*)d_in[29];
  const float* fc1_b  = (const float*)d_in[30];
  const float* fc2_w  = (const float*)d_in[31];
  const float* fc2_b  = (const float*)d_in[32];

  // --- f32 workspace region ---
  float* ws = (float*)d_ws;
  size_t off = 0;
  float* ffv   = ws + off; off += (size_t)B_ * CH * 64 * 512;    // 1,179,648
  float* fbev  = ws + off; off += (size_t)B_ * CH * 200 * 176;   // 1,267,200
  float* sfv   = ws + off; off += (size_t)NTOT * HID;            //   921,600
  float* sbev  = ws + off; off += (size_t)NTOT * HID;            //   921,600
  float* xflat = ws + off; off += (size_t)NTOT * HID;            //   921,600
  float* hbuf  = ws + off; off += (size_t)NTOT * 225;            //   460,800

  // --- bf16 packed-weight region (fragment layout, 512 bf16 per fragment) ---
  __bf16* pk = (__bf16*)(ws + off);
  size_t poff = 0;
  __bf16* w1fv_p  = pk + poff; poff += (size_t)(4*9*2)  * 512;   // 72 frags
  __bf16* w1bev_p = pk + poff; poff += (size_t)(16*9*2) * 512;   // 288 frags
  __bf16* w2fv_p  = pk + poff; poff += (size_t)2 * 512;
  __bf16* w2bev_p = pk + poff; poff += (size_t)2 * 512;
  __bf16* wq_p    = pk + poff; poff += (size_t)2 * 512;
  __bf16* wk_p    = pk + poff; poff += (size_t)2 * 512;
  __bf16* wv_p    = pk + poff; poff += (size_t)2 * 512;
  __bf16* wf_p    = pk + poff; poff += (size_t)2 * 512;
  __bf16* fc1_p   = pk + poff; poff += (size_t)(15*15) * 512;    // 225 frags

  // --- weight repack (tiny; runs every launch, deterministic) ---
  repack_conv1<<<4*9*2,  32, 0, stream>>>(fv_w1, 128, w1fv_p);
  repack_conv1<<<16*9*2, 32, 0, stream>>>(bev_w1, 512, w1bev_p);
  repack_dense<<<2,   32, 0, stream>>>(fv_w2, 18, 18, 2, w2fv_p);
  repack_dense<<<2,   32, 0, stream>>>(bev_w2, 18, 18, 2, w2bev_p);
  repack_dense<<<2,   32, 0, stream>>>(Wq, E_, E_, 2, wq_p);
  repack_dense<<<2,   32, 0, stream>>>(Wk, E_, E_, 2, wk_p);
  repack_dense<<<2,   32, 0, stream>>>(Wv, E_, E_, 2, wv_p);
  repack_dense<<<2,   32, 0, stream>>>(Wf, E_, E_, 2, wf_p);
  repack_dense<<<15*15, 32, 0, stream>>>(fc1_w, 225, HID, 15, fc1_p);

  // FV preprocessing: Cin=128, H=64, W=512
  preconv_kernel<<<B_ * 64 * ((512 + 15) / 16), 32, 0, stream>>>(
      feat_fv, 128, 64, 512,
      w1fv_p, fv_b1, fv_gamma, fv_beta, fv_mean, fv_var, w2fv_p, fv_b2, ffv);

  // BEV preprocessing: Cin=512, H=200, W=176
  preconv_kernel<<<B_ * 200 * ((176 + 15) / 16), 32, 0, stream>>>(
      feat_bev, 512, 200, 176,
      w1bev_p, bev_b1, bev_gamma, bev_beta, bev_mean, bev_var, w2bev_p, bev_b2, fbev);

  sample_kernel<<<(NTOT * E_ + 255) / 256, 256, 0, stream>>>(rois, ffv, fbev, sfv, sbev);

  classifier_kernel<<<NTOT, 32, 0, stream>>>(sbev, sfv, wq_p, bq, wk_p, bk, wv_p, bv,
                                             wf_p, bf_, conv_w, conv_b, xflat);

  fc1_kernel<<<(NTOT / 16) * 15, 32, 0, stream>>>(xflat, fc1_p, fc1_b, hbuf);

  fc2_kernel<<<(NTOT + 255) / 256, 256, 0, stream>>>(hbuf, fc2_w, fc2_b, (float*)d_out);
}
